// LeafLayer_43301860279039
// MI455X (gfx1250) — matrix operand order repaired and checked by reading, not compiled, not run on hardware
//
#include <hip/hip_runtime.h>

// ---------------------------------------------------------------------------
// CDNA5 / gfx1250: bf16 WMMA fused transformer layer with TDM + async-LDS
// staging.
// v = h@Wv+bv ; q = h@Wq+bq ; k = h@Wk+bk
// att = softmax(q k^T) ; out = att v ; hid = relu(out@W1+b1) ; y = hid@W2+b2
// ---------------------------------------------------------------------------

typedef __attribute__((ext_vector_type(16))) __bf16 v16bf;
typedef __attribute__((ext_vector_type(8)))  __bf16 v8bf;
typedef __attribute__((ext_vector_type(8)))  float  v8f;

static constexpr int Bb   = 16;
static constexpr int Nn   = 2048;
static constexpr int Hh   = 512;
static constexpr int ROWS = Bb * Nn;   // 32768

// ---------------- fp32 -> bf16 convert ----------------
__global__ void cvt_f32_bf16(const float* __restrict__ src,
                             __bf16* __restrict__ dst, int n) {
  int i = blockIdx.x * blockDim.x + threadIdx.x;
  int stride = gridDim.x * blockDim.x;
  for (; i < n; i += stride) dst[i] = (__bf16)src[i];
}

// 16x32 bf16 A-fragment: lanes 0-15: K=0..15 ;
// lanes 16-31: K=16..31
__device__ inline v16bf make_afrag(const __bf16* rowPtr, int kg) {
  v8bf lo = *(const v8bf*)(rowPtr + kg * 16);
  v8bf hi = *(const v8bf*)(rowPtr + kg * 16 + 8);
  return __builtin_shufflevector(lo, hi, 0,1,2,3,4,5,6,7,8,9,10,11,12,13,14,15);
}

__device__ inline v8f wmma_bf16(v16bf a, v16bf b, v8f c) {
  return __builtin_amdgcn_wmma_f32_16x16x32_bf16(false, a, false, b,
                                                 (short)0, c, false, false);
}

__device__ inline unsigned lds_offset(const void* p) {
  // flat LDS-aperture address: low 32 bits are the LDS byte offset
  return (unsigned)(unsigned long long)(uintptr_t)p;
}

// ---------------- Tensor Data Mover: 2-D bf16 tile -> LDS ----------------
#if __has_builtin(__builtin_amdgcn_tensor_load_to_lds)
#define USE_TDM 1
typedef unsigned int v4u __attribute__((ext_vector_type(4)));
typedef int          v8i __attribute__((ext_vector_type(8)));
typedef int          v4i __attribute__((ext_vector_type(4)));

// D# per ISA 8.3/8.4: group0 = {flags, lds_addr, global_addr, type=2},
// group1 = {data_size=2B, dims, tile dims, dim0 stride}. 2-D -> groups 2/3
// zero. This toolchain exposes the 6-arg builtin (g0, g1, g2, g3, g4, cpol).
__device__ inline void tdm_load_tile_2d(unsigned lds_addr,
                                        unsigned long long gaddr,
                                        unsigned tile_d0, unsigned tile_d1,
                                        unsigned stride0_elems) {
  v4u g0;
  g0[0] = 1u;                                       // count=1, user descriptor
  g0[1] = lds_addr;                                 // lds_addr[31:0]
  g0[2] = (unsigned)(gaddr & 0xFFFFFFFFu);          // global_addr[31:0]
  g0[3] = (unsigned)((gaddr >> 32) & 0x01FFFFFFu)   // global_addr[56:32]
          | (2u << 30);                             // type = 2 ("image")
  v8i g1;
  g1[0] = (int)(1u << 16);                          // data_size=1 -> 2 bytes
  g1[1] = (int)((tile_d0 & 0xFFFFu) << 16);         // tensor_dim0 = tile_d0
  g1[2] = (int)((tile_d0 >> 16) | ((tile_d1 & 0xFFFFu) << 16)); // tensor_dim1
  g1[3] = (int)((tile_d1 >> 16) | (tile_d0 << 16)); // tile_dim0
  g1[4] = (int)tile_d1;                             // tile_dim1 (tile_dim2=0)
  g1[5] = (int)stride0_elems;                       // tensor_dim0_stride[31:0]
  g1[6] = 0;
  g1[7] = 0;
  v4i z4 = {0, 0, 0, 0};
  v8i z8 = {0, 0, 0, 0, 0, 0, 0, 0};
  __builtin_amdgcn_tensor_load_to_lds(g0, g1, z4, z4, z8, 0);
}
#else
#define USE_TDM 0
#endif

// ---------------- generic bf16 GEMM: C = act(A @ W + bias) ----------------
// Block tile 128x128, 8 waves (4 row x 2 col), wave tile 32x64, BK=32.
// A tile staged by the Tensor Data Mover; W tile staged transposed by VALU.
template <bool RELU, bool OUTF32>
__global__ __launch_bounds__(256)
void gemm_bf16(const __bf16* __restrict__ A, const __bf16* __restrict__ W,
               const float* __restrict__ bias, void* __restrict__ Cout,
               int rows, int ncols, int K) {
  constexpr int BM = 128, BN = 128, BK = 32;
  __shared__ __bf16 As[BM][BK];   // [row][k]            8 KB
  __shared__ __bf16 Wt[BN][BK];   // [n][k] (transposed) 8 KB

  const int tid  = threadIdx.x;
  const int lane = tid & 31;
  const int wave = tid >> 5;
  const int waveRow = (wave & 3) * 32;
  const int waveCol = (wave >> 2) * 64;
  const int rowBase = blockIdx.x * BM;
  const int colBase = blockIdx.y * BN;
  const int kg = lane >> 4;

  v8f acc[2][4] = {};

  for (int kb = 0; kb < K; kb += BK) {
#if USE_TDM
    if (wave == 0) {
      tdm_load_tile_2d(lds_offset(&As[0][0]),
                       (unsigned long long)(uintptr_t)
                           (A + (size_t)rowBase * K + kb),
                       BK, BM, (unsigned)K);
    }
#else
    {
      int r = tid >> 1, hs = (tid & 1) * 16;
      const __bf16* src = A + (size_t)(rowBase + r) * K + kb + hs;
      v8bf d0 = *(const v8bf*)(src);
      v8bf d1 = *(const v8bf*)(src + 8);
      *(v8bf*)&As[r][hs]     = d0;
      *(v8bf*)&As[r][hs + 8] = d1;
    }
#endif
    // ---- stage W tile transposed: Wt[n][k] ----
    {
      int kk = tid >> 3, ng = (tid & 7) * 16;
      const __bf16* src = W + (size_t)(kb + kk) * ncols + colBase + ng;
      v8bf w0 = *(const v8bf*)(src);
      v8bf w1 = *(const v8bf*)(src + 8);
      if (kb + BK < K)   // prefetch next W tile into GL2
        __builtin_prefetch(src + (size_t)BK * ncols, 0, 1);
      #pragma unroll
      for (int j = 0; j < 8; ++j) Wt[ng + j][kk]     = w0[j];
      #pragma unroll
      for (int j = 0; j < 8; ++j) Wt[ng + 8 + j][kk] = w1[j];
    }
#if USE_TDM
    if (wave == 0) asm volatile("s_wait_tensorcnt 0x0" ::: "memory");
#endif
    __syncthreads();

    v16bf af[2], bf[4];
    #pragma unroll
    for (int rt = 0; rt < 2; ++rt)
      af[rt] = make_afrag(&As[waveRow + rt * 16 + (lane & 15)][0], kg);
    #pragma unroll
    for (int ct = 0; ct < 4; ++ct)
      bf[ct] = *(const v16bf*)&Wt[waveCol + ct * 16 + (lane & 15)][kg * 16];
    #pragma unroll
    for (int rt = 0; rt < 2; ++rt)
      #pragma unroll
      for (int ct = 0; ct < 4; ++ct)
        acc[rt][ct] = wmma_bf16(af[rt], bf[ct], acc[rt][ct]);
    __syncthreads();
  }

  // ---- epilogue: bias (+relu). C layout: VGPR r -> rows r / r+8 ----
  #pragma unroll
  for (int rt = 0; rt < 2; ++rt) {
    #pragma unroll
    for (int ct = 0; ct < 4; ++ct) {
      int n = colBase + waveCol + ct * 16 + (lane & 15);
      float bv = bias[n];
      int rb = rowBase + waveRow + rt * 16 + ((lane & 16) ? 8 : 0);
      #pragma unroll
      for (int r = 0; r < 8; ++r) {
        float val = acc[rt][ct][r] + bv;
        if (RELU) val = fmaxf(val, 0.0f);
        size_t idx = (size_t)(rb + r) * ncols + n;
        if (OUTF32) ((float*)Cout)[idx] = val;
        else        ((__bf16*)Cout)[idx] = (__bf16)val;
      }
    }
  }
}

// ---------------- fused flash attention ----------------
// WG = 4 waves; each wave owns 16 query rows with a full 16x512 f32
// accumulator (32 WMMA tiles, 256 VGPRs) + all 16 q A-fragments preloaded.
// Per 32-key block: K tile async-copied to LDS (linear 32 KB), V tile staged
// transposed; scores -> online softmax -> P@V, all in-WGP.
__global__ __launch_bounds__(128)
void attn_flash(const __bf16* __restrict__ q, const __bf16* __restrict__ k,
                const __bf16* __restrict__ v, __bf16* __restrict__ out) {
  __shared__ __bf16 ks[32][Hh];       // [key][h]  32 KB
  __shared__ __bf16 vts[Hh][32];      // [h][key]  32 KB
  __shared__ __bf16 ps[4][16][32];    // per-wave P tile, 4 KB

  const int tid  = threadIdx.x;
  const int lane = tid & 31;
  const int wave = tid >> 5;
  const int kg   = lane >> 4;
  const int b    = blockIdx.x >> 5;
  const int rb   = (blockIdx.x & 31) * 64;
  const int myRow0 = rb + wave * 16;

  const __bf16* qb  = q + (size_t)b * Nn * Hh;
  const __bf16* kbm = k + (size_t)b * Nn * Hh;
  const __bf16* vb  = v + (size_t)b * Nn * Hh;

  // preload all q A-fragments for this wave's 16 rows (key-block invariant)
  const __bf16* qrow = qb + (size_t)(myRow0 + (lane & 15)) * Hh;
  v16bf qa[16];
  #pragma unroll
  for (int i = 0; i < 16; ++i) qa[i] = make_afrag(qrow + i * 32, kg);

  v8f o[32] = {};
  float mrow[8], lrow[8];
  #pragma unroll
  for (int r = 0; r < 8; ++r) { mrow[r] = -1e30f; lrow[r] = 0.0f; }

  const unsigned ks_base = lds_offset(&ks[0][0]);

  for (int kt = 0; kt < Nn; kt += 32) {
    // ---- async copy K tile (rows kt..kt+31 are contiguous: linear 32 KB) --
    {
      const __bf16* ksrc = kbm + (size_t)kt * Hh;
      #pragma unroll
      for (int i = 0; i < 16; ++i) {
        int f = tid + i * 128;                       // 2048 16-byte chunks
        unsigned la = ks_base + (unsigned)f * 16u;
        unsigned long long ga =
            (unsigned long long)(uintptr_t)(ksrc + (size_t)f * 8);
        asm volatile("global_load_async_to_lds_b128 %0, %1, off"
                     :: "v"(la), "v"(ga) : "memory");
      }
    }
    // ---- stage V^T tile: vts[h][key] ----
    #pragma unroll
    for (int i = 0; i < 16; ++i) {
      int f   = tid + i * 128;
      int key = f >> 6;
      int h0  = (f & 63) * 8;
      v8bf d = *(const v8bf*)(vb + (size_t)(kt + key) * Hh + h0);
      #pragma unroll
      for (int j = 0; j < 8; ++j) vts[h0 + j][key] = d[j];
    }
    asm volatile("s_wait_asynccnt 0x0" ::: "memory");
    __syncthreads();

    // ---- scores S = q k^T (two 16x16 tiles) from LDS ----
    v8f s0 = {}, s1 = {};
    const __bf16* k0 = &ks[lane & 15][kg * 16];
    const __bf16* k1 = &ks[16 + (lane & 15)][kg * 16];
    #pragma unroll
    for (int i = 0; i < 16; ++i) {
      s0 = wmma_bf16(qa[i], *(const v16bf*)(k0 + i * 32), s0);
      s1 = wmma_bf16(qa[i], *(const v16bf*)(k1 + i * 32), s1);
    }

    // ---- online softmax (16-lane-half row reductions) ----
    float alpha[8];
    #pragma unroll
    for (int r = 0; r < 8; ++r) {
      float vmax = fmaxf(s0[r], s1[r]);
      #pragma unroll
      for (int msk = 8; msk >= 1; msk >>= 1)
        vmax = fmaxf(vmax, __shfl_xor(vmax, msk, 32));
      float mn = fmaxf(mrow[r], vmax);
      float al = __expf(mrow[r] - mn);
      float p0 = __expf(s0[r] - mn);
      float p1 = __expf(s1[r] - mn);
      float rs = p0 + p1;
      #pragma unroll
      for (int msk = 8; msk >= 1; msk >>= 1)
        rs += __shfl_xor(rs, msk, 32);
      lrow[r] = lrow[r] * al + rs;
      mrow[r] = mn;
      alpha[r] = al;
      int prow = r + ((lane & 16) ? 8 : 0);
      ps[wave][prow][(lane & 15)]      = (__bf16)p0;
      ps[wave][prow][(lane & 15) + 16] = (__bf16)p1;
    }
    #pragma unroll
    for (int t = 0; t < 32; ++t)
      #pragma unroll
      for (int r = 0; r < 8; ++r) o[t][r] *= alpha[r];

    // ---- out += P(16x32) @ V(32x512) ----
    v16bf pa = make_afrag(&ps[wave][lane & 15][0], kg);
    #pragma unroll
    for (int t = 0; t < 32; ++t) {
      v16bf bfr = *(const v16bf*)&vts[t * 16 + (lane & 15)][kg * 16];
      o[t] = wmma_bf16(pa, bfr, o[t]);
    }
    __syncthreads();
  }

  // ---- normalize and store bf16 ----
  #pragma unroll
  for (int r = 0; r < 8; ++r) lrow[r] = 1.0f / lrow[r];
  __bf16* ob = out + ((size_t)b * Nn + myRow0) * Hh;
  #pragma unroll
  for (int t = 0; t < 32; ++t) {
    int n = t * 16 + (lane & 15);
    int roff = (lane & 16) ? 8 : 0;
    #pragma unroll
    for (int r = 0; r < 8; ++r)
      ob[(size_t)(r + roff) * Hh + n] = (__bf16)(o[t][r] * lrow[r]);
  }
}

// ---------------- driver ----------------
extern "C" void kernel_launch(void* const* d_in, const int* in_sizes, int n_in,
                              void* d_out, int out_size, void* d_ws, size_t ws_size,
                              hipStream_t stream) {
  // inputs: 0=d(unused) 1=h 2=Wv 3=bv 4=Wk 5=bk 6=Wq 7=bq 8=W1 9=b1 10=W2 11=b2
  const float* h  = (const float*)d_in[1];
  const float* Wv = (const float*)d_in[2];  const float* bv = (const float*)d_in[3];
  const float* Wk = (const float*)d_in[4];  const float* bk = (const float*)d_in[5];
  const float* Wq = (const float*)d_in[6];  const float* bq = (const float*)d_in[7];
  const float* W1 = (const float*)d_in[8];  const float* b1 = (const float*)d_in[9];
  const float* W2 = (const float*)d_in[10]; const float* b2 = (const float*)d_in[11];

  char* ws = (char*)d_ws;
  const size_t ACT = (size_t)ROWS * Hh * sizeof(__bf16);   // 32 MB
  const size_t WSZ = (size_t)Hh * Hh * sizeof(__bf16);     // 512 KB
  __bf16* h_bf   = (__bf16*)(ws);
  __bf16* q_bf   = (__bf16*)(ws + ACT);
  __bf16* k_bf   = (__bf16*)(ws + 2 * ACT);
  __bf16* v_bf   = (__bf16*)(ws + 3 * ACT);
  __bf16* o_bf   = (__bf16*)(ws + 4 * ACT);
  __bf16* hid_bf = (__bf16*)(ws + 5 * ACT);
  __bf16* Wq_bf  = (__bf16*)(ws + 6 * ACT);
  __bf16* Wk_bf  = (__bf16*)(ws + 6 * ACT + WSZ);
  __bf16* Wv_bf  = (__bf16*)(ws + 6 * ACT + 2 * WSZ);
  __bf16* W1_bf  = (__bf16*)(ws + 6 * ACT + 3 * WSZ);
  __bf16* W2_bf  = (__bf16*)(ws + 6 * ACT + 4 * WSZ);

  const int nAct = ROWS * Hh;
  const int nW   = Hh * Hh;
  cvt_f32_bf16<<<4096, 256, 0, stream>>>(h,  h_bf,  nAct);
  cvt_f32_bf16<<<256,  256, 0, stream>>>(Wq, Wq_bf, nW);
  cvt_f32_bf16<<<256,  256, 0, stream>>>(Wk, Wk_bf, nW);
  cvt_f32_bf16<<<256,  256, 0, stream>>>(Wv, Wv_bf, nW);
  cvt_f32_bf16<<<256,  256, 0, stream>>>(W1, W1_bf, nW);
  cvt_f32_bf16<<<256,  256, 0, stream>>>(W2, W2_bf, nW);

  dim3 gg(ROWS / 128, Hh / 128);
  gemm_bf16<false, false><<<gg, 256, 0, stream>>>(h_bf, Wq_bf, bq, q_bf, ROWS, Hh, Hh);
  gemm_bf16<false, false><<<gg, 256, 0, stream>>>(h_bf, Wk_bf, bk, k_bf, ROWS, Hh, Hh);
  gemm_bf16<false, false><<<gg, 256, 0, stream>>>(h_bf, Wv_bf, bv, v_bf, ROWS, Hh, Hh);

  attn_flash<<<Bb * (Nn / 64), 128, 0, stream>>>(q_bf, k_bf, v_bf, o_bf);

  gemm_bf16<true,  false><<<gg, 256, 0, stream>>>(o_bf,   W1_bf, b1, hid_bf, ROWS, Hh, Hh);
  gemm_bf16<false, true ><<<gg, 256, 0, stream>>>(hid_bf, W2_bf, b2, d_out,  ROWS, Hh, Hh);
}
